// BigST_16939351015450
// MI455X (gfx1250) — compile-verified, matrix operand order checked by reading, not run on hardware
//
#include <hip/hip_runtime.h>

typedef unsigned short u16;
typedef unsigned int   u32;

typedef __attribute__((ext_vector_type(16))) __bf16 v16bf;
typedef __attribute__((ext_vector_type(8)))  float  v8f;
typedef __attribute__((ext_vector_type(4)))  unsigned int u32x4;
typedef __attribute__((ext_vector_type(8)))  int i32x8;
typedef __attribute__((ext_vector_type(4)))  int i32x4;

#define BN   131072     // B*N
#define NN   8192       // N
#define BB   16         // B
#define CC   192        // channels
#define MM   64         // random features

union FragB { v16bf v; u16 us[16]; uint4 q[2]; };

__device__ __forceinline__ u16 f2bf(float f) {
  union { float f; u32 u; } c; c.f = f;
  u32 r = (c.u + 0x7fffu + ((c.u >> 16) & 1u)) >> 16;
  return (u16)r;
}
__device__ __forceinline__ float bf2f(u16 h) {
  union { u32 u; float f; } c; c.u = ((u32)h) << 16; return c.f;
}
__device__ __forceinline__ u32 encf(float f) {
  union { float f; u32 u; } c; c.f = f;
  return (c.u & 0x80000000u) ? ~c.u : (c.u | 0x80000000u);
}
__device__ __forceinline__ float decf(u32 e) {
  union { u32 u; float f; } c;
  c.u = (e & 0x80000000u) ? (e & 0x7fffffffu) : ~e;
  return c.f;
}

// Load one 16x32 bf16 A/B fragment from a row-major [rows x stride] u16 matrix.
// A layout (ISA 7.12.2): lanes 0-15 row=row0+lane, elems = K{k0..k0+7, k0+16..k0+23};
// lanes 16-31 row=row0+lane-16, elems = K{k0+8..k0+15, k0+24..k0+31}.
// B fragment uses the same pattern with "row" = output column (weight stored row-major [out][k]).
__device__ __forceinline__ v16bf load_frag(const u16* base, size_t row0, int stride, int k0) {
  int lane = threadIdx.x & 31;
  int half = lane >> 4;
  size_t r = row0 + (size_t)(lane & 15);
  const u16* p = base + r * (size_t)stride + k0 + half * 8;
  FragB f;
  f.q[0] = *(const uint4*)(p);
  f.q[1] = *(const uint4*)(p + 16);
  return f.v;
}

__device__ __forceinline__ v16bf load_frag_relu(const u16* base, size_t row0, int stride, int k0) {
  int lane = threadIdx.x & 31;
  int half = lane >> 4;
  size_t r = row0 + (size_t)(lane & 15);
  const u16* p = base + r * (size_t)stride + k0 + half * 8;
  FragB f;
  f.q[0] = *(const uint4*)(p);
  f.q[1] = *(const uint4*)(p + 16);
#pragma unroll
  for (int i = 0; i < 16; ++i)
    if (f.us[i] & 0x8000u) f.us[i] = 0;   // bf16 relu: negatives -> +0
  return f.v;
}

// Gather a fragment where the K dimension is the "node" index of an LDS tile
// lds layout: [node][ldstride], fixed = per-lane row/col index.
__device__ __forceinline__ v16bf gather_frag(const u16* lds, int fixed, int ldstride, int n0) {
  int lane = threadIdx.x & 31;
  int half = lane >> 4;
  FragB f;
#pragma unroll
  for (int e = 0; e < 16; ++e) {
    int node = n0 + (e & 7) + ((e >> 3) << 4) + half * 8;
    f.us[e] = lds[node * ldstride + fixed];
  }
  return f.v;
}

__device__ __forceinline__ v8f wmma_bf16(v16bf a, v16bf b, v8f c) {
  return __builtin_amdgcn_wmma_f32_16x16x32_bf16(false, a, false, b, (short)0, c,
                                                 false, false);
}
__device__ __forceinline__ v8f vzero8() {
  v8f z = {0.f, 0.f, 0.f, 0.f, 0.f, 0.f, 0.f, 0.f};
  return z;
}

// TDM: DMA a contiguous [nbytes] block global -> LDS as a 1-row tensor tile
// (data_size=8B, tile_dim0=nbytes/8, tile_dim1=1). Descriptor per ISA 08 §8.
__device__ __forceinline__ void tdm_load(u32 lds_addr, const void* gaddr, u32 nbytes) {
  unsigned long long ga = (unsigned long long)gaddr;
  u32 n8 = nbytes >> 3;
  u32x4 g0;
  g0.x = 1u;                                            // count=1, user mode
  g0.y = lds_addr;                                      // lds byte address
  g0.z = (u32)(ga & 0xffffffffu);                       // global addr lo
  g0.w = (u32)((ga >> 32) & 0x01ffffffu) | (2u << 30);  // addr hi | type=2
  i32x8 g1;
  g1[0] = (int)(3u << 16);        // wg_mask=0, data_size=8B
  g1[1] = (int)(n8 << 16);        // tensor_dim0[15:0] in [31:16]
  g1[2] = (int)(1u << 16);        // tensor_dim0 hi=0, tensor_dim1=1
  g1[3] = (int)(n8 << 16);        // tensor_dim1 hi=0, tile_dim0=n8
  g1[4] = 1;                      // tile_dim1=1, tile_dim2=0
  g1[5] = (int)n8;                // tensor_dim0_stride lo
  g1[6] = 0;
  g1[7] = 0;
  i32x4 z4 = {0, 0, 0, 0};
#if defined(__clang_major__) && (__clang_major__ >= 23)
  i32x8 z8 = {0, 0, 0, 0, 0, 0, 0, 0};
  __builtin_amdgcn_tensor_load_to_lds(g0, g1, z4, z4, z8, 0);
#else
  __builtin_amdgcn_tensor_load_to_lds(g0, g1, z4, z4, 0);
#endif
}

// ---------------------------------------------------------------- prep kernels
__global__ void k_conv(const float* __restrict__ in, u16* __restrict__ out, int n) {
  for (int i = blockIdx.x * blockDim.x + threadIdx.x; i < n; i += gridDim.x * blockDim.x)
    out[i] = f2bf(in[i]);
}

__global__ void k_zero(float* __restrict__ p, int n) {
  for (int i = blockIdx.x * blockDim.x + threadIdx.x; i < n; i += gridDim.x * blockDim.x)
    p[i] = 0.f;
}

__global__ void k_init(float* __restrict__ v2s, u32* __restrict__ kmax) {
  int t = threadIdx.x;
  if (t < BB * MM) v2s[t] = 0.f;
  if (t < BB) kmax[t] = 0u;
}

// W_in (64x36) -> padded bf16 64x64
__global__ void k_prep_win(const float* __restrict__ W, u16* __restrict__ Wb) {
  int t = blockIdx.x * blockDim.x + threadIdx.x;
  if (t >= 64 * 64) return;
  int r = t >> 6, k = t & 63;
  Wb[t] = (k < 36) ? f2bf(W[r * 36 + k]) : (u16)0;
}

// Wcat (256x128) = [W1; W2; s*proj@W1; s*proj@W2], bcat likewise
__global__ void k_prep_wcat(const float* __restrict__ W1, const float* __restrict__ W2,
                            const float* __restrict__ proj,
                            const float* __restrict__ b1, const float* __restrict__ b2,
                            u16* __restrict__ Wcatb, float* __restrict__ bcat) {
  const float s = 0.3535533905932738f;  // 64^-0.25 * tau^-0.5
  int t = blockIdx.x * blockDim.x + threadIdx.x;
  if (t >= 256 * 128) return;
  int r = t >> 7, k = t & 127;
  float v;
  if (r < 64)       v = W1[r * 128 + k];
  else if (r < 128) v = W2[(r - 64) * 128 + k];
  else if (r < 192) {
    int rr = r - 128; float a = 0.f;
    for (int j = 0; j < 64; ++j) a += proj[rr * 64 + j] * W1[j * 128 + k];
    v = s * a;
  } else {
    int rr = r - 192; float a = 0.f;
    for (int j = 0; j < 64; ++j) a += proj[rr * 64 + j] * W2[j * 128 + k];
    v = s * a;
  }
  Wcatb[t] = f2bf(v);
  if (k == 0) {
    float bv;
    if (r < 64)       bv = b1[r];
    else if (r < 128) bv = b2[r - 64];
    else if (r < 192) {
      int rr = r - 128; float a = 0.f;
      for (int j = 0; j < 64; ++j) a += proj[rr * 64 + j] * b1[j];
      bv = s * a;
    } else {
      int rr = r - 192; float a = 0.f;
      for (int j = 0; j < 64; ++j) a += proj[rr * 64 + j] * b2[j];
      bv = s * a;
    }
    bcat[r] = bv;
  }
}

// W_reg (12x384) -> padded bf16 16x384
__global__ void k_prep_wreg(const float* __restrict__ W, u16* __restrict__ Wb) {
  int t = blockIdx.x * blockDim.x + threadIdx.x;
  if (t >= 16 * 384) return;
  int r = t / 384, k = t % 384;
  Wb[t] = (r < 12) ? f2bf(W[r * 384 + k]) : (u16)0;
}

// ------------------------------------------------------------- gather kernel
__global__ __launch_bounds__(256) void k_gather(
    const float* __restrict__ x, const float* __restrict__ node_emb,
    const float* __restrict__ time_emb,
    u16* __restrict__ xrb, u16* __restrict__ ntb,
    float* __restrict__ hf, u16* __restrict__ h0b, u16* __restrict__ hb) {
  size_t g = (size_t)blockIdx.x * blockDim.x + threadIdx.x;
  if (g >= BN) return;
  int n = (int)(g & (NN - 1));
  const float* xp = x + g * 36;
  float tv = xp[11 * 3 + 1];
  int ti = (int)(tv * 288.0f);
  ti = ti < 0 ? 0 : (ti > 287 ? 287 : ti);
  const float* ne = node_emb + (size_t)n * 64;
  const float* te = time_emb + (size_t)ti * 64;
  u16* xr = xrb + g * 64;
#pragma unroll
  for (int j = 0; j < 36; ++j) xr[j] = f2bf(xp[j]);
#pragma unroll
  for (int j = 36; j < 64; ++j) xr[j] = 0;
  u16* nt = ntb + g * 128;
  float* hrow = hf + g * CC;
  u16* h0r = h0b + g * CC;
  u16* hr = hb + g * CC;
#pragma unroll 8
  for (int j = 0; j < 64; ++j) {
    float v = ne[j]; u16 hv = f2bf(v);
    nt[j] = hv; hrow[64 + j] = v; h0r[64 + j] = hv; hr[64 + j] = hv;
  }
#pragma unroll 8
  for (int j = 0; j < 64; ++j) {
    float v = te[j]; u16 hv = f2bf(v);
    nt[64 + j] = hv; hrow[128 + j] = v; h0r[128 + j] = hv; hr[128 + j] = hv;
  }
}

// ------------------------------------------------------ input embedding GEMM
__global__ __launch_bounds__(256) void k_inemb(
    const u16* __restrict__ xrb, const u16* __restrict__ Winb,
    const float* __restrict__ b_in,
    float* __restrict__ hf, u16* __restrict__ h0b, u16* __restrict__ hb) {
  int wave = threadIdx.x >> 5, lane = threadIdx.x & 31, half = lane >> 4;
  size_t g0 = ((size_t)blockIdx.x * 8 + wave) * 16;
  v16bf a0 = load_frag(xrb, g0, 64, 0);
  v16bf a1 = load_frag(xrb, g0, 64, 32);
#pragma unroll
  for (int ct = 0; ct < 4; ++ct) {
    v8f acc = vzero8();
    acc = wmma_bf16(a0, load_frag(Winb, ct * 16, 64, 0), acc);
    acc = wmma_bf16(a1, load_frag(Winb, ct * 16, 64, 32), acc);
    int c = ct * 16 + (lane & 15);
    float bv = b_in[c];
#pragma unroll
    for (int r = 0; r < 8; ++r) {
      size_t g = g0 + r + half * 8;
      float v = acc[r] + bv;
      u16 hv = f2bf(v);
      hf[g * CC + c] = v;
      h0b[g * CC + c] = hv;
      hb[g * CC + c] = hv;
    }
  }
}

// --------------------------------- fused nv1/nv2/dd1/dd2 GEMM + FAVOR+ epilogue
__global__ __launch_bounds__(128) void k_nvdd(
    const u16* __restrict__ ntb, const u16* __restrict__ Wcatb,
    const float* __restrict__ bcat,
    u16* __restrict__ v1pb, float* __restrict__ kd, u32* __restrict__ kmaxInt) {
  __shared__ float Cst[4][16][256];  // 64KB
  int wave = threadIdx.x >> 5, lane = threadIdx.x & 31, half = lane >> 4;
  size_t g0 = ((size_t)blockIdx.x * 4 + wave) * 16;
  v16bf aF[4];
#pragma unroll
  for (int ks = 0; ks < 4; ++ks) aF[ks] = load_frag(ntb, g0, 128, ks * 32);
#pragma unroll
  for (int ct = 0; ct < 16; ++ct) {
    v8f acc = vzero8();
#pragma unroll
    for (int ks = 0; ks < 4; ++ks)
      acc = wmma_bf16(aF[ks], load_frag(Wcatb, ct * 16, 128, ks * 32), acc);
    int c = ct * 16 + (lane & 15);
    float bv = bcat[c];
#pragma unroll
    for (int r = 0; r < 8; ++r) Cst[wave][r + half * 8][c] = acc[r] + bv;
  }
  __syncthreads();
  {
    // lane pair (row, row+16) splits the 64-feature range
    int row = lane & 15, part = half;
    size_t g = g0 + row;
    int b = (int)(g >> 13);  // N = 8192
    const float* Z = Cst[wave][row];
    const float s = 0.3535533905932738f;
    float ss1 = 0.f, ss2 = 0.f, mx1 = -3.0e38f, mx2 = -3.0e38f;
#pragma unroll 8
    for (int m = part * 32; m < part * 32 + 32; ++m) {
      ss1 += Z[m] * Z[m];
      ss2 += Z[64 + m] * Z[64 + m];
      mx1 = fmaxf(mx1, Z[128 + m]);
      mx2 = fmaxf(mx2, Z[192 + m]);
    }
    ss1 += __shfl_xor(ss1, 16, 32);
    ss2 += __shfl_xor(ss2, 16, 32);
    mx1 = fmaxf(mx1, __shfl_xor(mx1, 16, 32));
    mx2 = fmaxf(mx2, __shfl_xor(mx2, 16, 32));
    float diag1 = 0.5f * s * s * ss1;
    float diag2 = 0.5f * s * s * ss2;
#pragma unroll 8
    for (int m = part * 32; m < part * 32 + 32; ++m) {
      v1pb[g * MM + m] = f2bf(0.125f * (__expf(Z[128 + m] - diag1 - mx1) + 1e-6f));
      kd[g * MM + m] = Z[192 + m] - diag2;
    }
    if (part == 0) atomicMax(kmaxInt + b, encf(mx2));
  }
}

// -------------------------------------------------- key feature map + v2s sum
__global__ __launch_bounds__(256) void k_v2p(
    const float* __restrict__ kd, const u32* __restrict__ kmaxInt,
    u16* __restrict__ v2pb, float* __restrict__ v2s) {
  int b = blockIdx.y, chunk = blockIdx.x;
  int m = threadIdx.x & 63, sub = threadIdx.x >> 6;
  float kmax = decf(kmaxInt[b]);
  float part = 0.f;
  size_t gbase = (size_t)b * NN + (size_t)chunk * 256 + (size_t)sub * 64;
  for (int i = 0; i < 64; ++i) {
    size_t g = gbase + i;
    float v = 0.125f * (__expf(kd[g * MM + m] - kmax) + 1e-6f);
    v2pb[g * MM + m] = f2bf(v);
    part += v;
  }
  __shared__ float red[256];
  red[threadIdx.x] = part;
  __syncthreads();
  if (sub == 0) {
    float t = red[m] + red[64 + m] + red[128 + m] + red[192 + m];
    atomicAdd(&v2s[b * MM + m], t);
  }
}

// --------------------------------------------------------------- gate GEMM
__global__ __launch_bounds__(256) void k_gate(
    const u16* __restrict__ hb, const u16* __restrict__ WiB,
    const u16* __restrict__ WoB, const float* __restrict__ biL,
    const float* __restrict__ boL, u16* __restrict__ gb) {
  if (threadIdx.x == 0) { __builtin_prefetch(WiB, 0, 1); __builtin_prefetch(WoB, 0, 1); }
  int wave = threadIdx.x >> 5, lane = threadIdx.x & 31, half = lane >> 4;
  size_t g0 = ((size_t)blockIdx.x * 8 + wave) * 16;
  v16bf aF[6];
#pragma unroll
  for (int ks = 0; ks < 6; ++ks) aF[ks] = load_frag(hb, g0, CC, ks * 32);
#pragma unroll 2
  for (int p = 0; p < 12; ++p) {
    v8f ai = vzero8(), ao = vzero8();
#pragma unroll
    for (int ks = 0; ks < 6; ++ks) {
      ai = wmma_bf16(aF[ks], load_frag(WiB, p * 16, CC, ks * 32), ai);
      ao = wmma_bf16(aF[ks], load_frag(WoB, p * 16, CC, ks * 32), ao);
    }
    int c = p * 16 + (lane & 15);
    float bi_v = biL[c], bo_v = boL[c];
#pragma unroll
    for (int r = 0; r < 8; ++r) {
      size_t g = g0 + r + half * 8;
      float wi = ai[r] + bi_v, wo = ao[r] + bo_v;
      float gv = wo / (1.f + __expf(-wi));
      gb[g * CC + c] = f2bf(gv);
    }
  }
}

// ------------------------------------------- v2x = sum_n v2p^T * g (split-K)
// Tiles staged into LDS by the Tensor Data Mover (async DMA, TENSORcnt).
__global__ __launch_bounds__(256) void k_v2x(
    const u16* __restrict__ v2pb, const u16* __restrict__ gb,
    float* __restrict__ v2x) {
  __shared__ u16 sv[128 * 64];    // 16KB
  __shared__ u16 sg[128 * 192];   // 48KB
  int b = blockIdx.y, split = blockIdx.x;
  size_t gbase = (size_t)b * NN + (size_t)split * 128;
  if (threadIdx.x < 32) {              // wave 0 issues both DMAs
    if (threadIdx.x == 0) {
      tdm_load((u32)(size_t)(void*)sv, v2pb + gbase * MM, 128 * 64 * 2);
      tdm_load((u32)(size_t)(void*)sg, gb + gbase * CC, 128 * 192 * 2);
    }
    __builtin_amdgcn_s_wait_tensorcnt(0);
  }
  __syncthreads();
  int wave = threadIdx.x >> 5, lane = threadIdx.x & 31, half = lane >> 4;
  for (int job = wave; job < 48; job += 8) {
    int rt = job / 12, ct = job % 12;
    int m = rt * 16 + (lane & 15);
    int c = ct * 16 + (lane & 15);
    v8f acc = vzero8();
#pragma unroll
    for (int ks = 0; ks < 4; ++ks) {
      v16bf a = gather_frag(sv, m, 64, ks * 32);
      v16bf bb = gather_frag(sg, c, CC, ks * 32);
      acc = wmma_bf16(a, bb, acc);
    }
#pragma unroll
    for (int r = 0; r < 8; ++r) {
      int mr = rt * 16 + r + half * 8;
      atomicAdd(&v2x[((size_t)b * MM + mr) * CC + c], acc[r]);
    }
  }
}

// ---------------------------------------------- v2x f32 -> transposed bf16
__global__ void k_trans(const float* __restrict__ v2x, u16* __restrict__ v2xbT) {
  int total = BB * MM * CC;
  for (int i = blockIdx.x * blockDim.x + threadIdx.x; i < total;
       i += gridDim.x * blockDim.x) {
    int b = i / (MM * CC);
    int rem = i - b * (MM * CC);
    int m = rem / CC, c = rem % CC;
    v2xbT[((size_t)b * CC + c) * MM + m] = f2bf(v2x[i]);
  }
}

// ----------------------- out1 = v1p @ v2x ; /out2 ; +res ; LayerNorm ; write
__global__ __launch_bounds__(128) void k_attnout(
    const u16* __restrict__ v1pb, const u16* __restrict__ v2xbT,
    const float* __restrict__ v2s, const float* __restrict__ lng,
    const float* __restrict__ lnb, float* __restrict__ hf, u16* __restrict__ hb) {
  __shared__ float Cst[4][16][CC];  // 48KB
  int wave = threadIdx.x >> 5, lane = threadIdx.x & 31, half = lane >> 4;
  size_t g0 = ((size_t)blockIdx.x * 4 + wave) * 16;
  int b = (int)(g0 >> 13);
  v16bf a0 = load_frag(v1pb, g0, 64, 0);
  v16bf a1 = load_frag(v1pb, g0, 64, 32);
  const u16* Bm = v2xbT + (size_t)b * CC * MM;
#pragma unroll
  for (int ct = 0; ct < 12; ++ct) {
    v8f acc = vzero8();
    acc = wmma_bf16(a0, load_frag(Bm, ct * 16, 64, 0), acc);
    acc = wmma_bf16(a1, load_frag(Bm, ct * 16, 64, 32), acc);
    int c = ct * 16 + (lane & 15);
#pragma unroll
    for (int r = 0; r < 8; ++r) Cst[wave][r + half * 8][c] = acc[r];
  }
  __syncthreads();
  {
    // lane pair (row, row+16) splits features/channels, combined via shfl_xor
    int row = lane & 15, part = half;
    size_t g = g0 + row;
    float out2 = 0.f;
#pragma unroll 8
    for (int m = part * 32; m < part * 32 + 32; ++m)
      out2 += bf2f(v1pb[g * MM + m]) * v2s[b * MM + m];
    out2 += __shfl_xor(out2, 16, 32);
    float inv = 1.f / out2;
    float sum = 0.f, sq = 0.f;
    float* Crow = Cst[wave][row];
    float* hrow = hf + g * CC;
#pragma unroll 8
    for (int c = part * 96; c < part * 96 + 96; ++c) {
      float hv = Crow[c] * inv + hrow[c];
      Crow[c] = hv;
      sum += hv; sq += hv * hv;
    }
    sum += __shfl_xor(sum, 16, 32);
    sq  += __shfl_xor(sq, 16, 32);
    float mu = sum * (1.f / CC);
    float var = sq * (1.f / CC) - mu * mu;
    float rstd = rsqrtf(var + 1e-5f);
    u16* hbr = hb + g * CC;
#pragma unroll 8
    for (int c = part * 96; c < part * 96 + 96; ++c) {
      float y = (Crow[c] - mu) * rstd * lng[c] + lnb[c];
      hrow[c] = y;
      hbr[c] = f2bf(y);
    }
  }
}

// ------------------------------------------------ regression head (K=384)
__global__ __launch_bounds__(256) void k_head(
    const u16* __restrict__ h0b, const u16* __restrict__ hb,
    const u16* __restrict__ WregB, const float* __restrict__ b_reg,
    float* __restrict__ outp) {
  if (threadIdx.x == 0) __builtin_prefetch(WregB, 0, 1);
  int wave = threadIdx.x >> 5, lane = threadIdx.x & 31, half = lane >> 4;
  size_t g0 = ((size_t)blockIdx.x * 8 + wave) * 16;
  v8f acc = vzero8();
#pragma unroll
  for (int ks = 0; ks < 6; ++ks)
    acc = wmma_bf16(load_frag_relu(h0b, g0, CC, ks * 32),
                    load_frag(WregB, 0, 384, ks * 32), acc);
#pragma unroll
  for (int ks = 0; ks < 6; ++ks)
    acc = wmma_bf16(load_frag_relu(hb, g0, CC, ks * 32),
                    load_frag(WregB, 0, 384, 192 + ks * 32), acc);
  int o = lane & 15;
  if (o < 12) {
    float bv = b_reg[o];
#pragma unroll
    for (int r = 0; r < 8; ++r) {
      size_t g = g0 + r + half * 8;
      size_t b = g >> 13;
      size_t n = g & (NN - 1);
      outp[(b * 12 + o) * NN + n] = acc[r] + bv;
    }
  }
}

// ---------------------------------------------------------------- launcher
extern "C" void kernel_launch(void* const* d_in, const int* in_sizes, int n_in,
                              void* d_out, int out_size, void* d_ws, size_t ws_size,
                              hipStream_t stream) {
  const float* x        = (const float*)d_in[0];
  const float* node_emb = (const float*)d_in[1];
  const float* time_emb = (const float*)d_in[2];
  const float* W_in     = (const float*)d_in[3];
  const float* b_in     = (const float*)d_in[4];
  const float* W1       = (const float*)d_in[5];
  const float* b1       = (const float*)d_in[6];
  const float* W2       = (const float*)d_in[7];
  const float* b2       = (const float*)d_in[8];
  const float* Wi       = (const float*)d_in[9];
  const float* bi       = (const float*)d_in[10];
  const float* Wo       = (const float*)d_in[11];
  const float* bo       = (const float*)d_in[12];
  const float* ln_g     = (const float*)d_in[13];
  const float* ln_b     = (const float*)d_in[14];
  const float* W_reg    = (const float*)d_in[15];
  const float* b_reg    = (const float*)d_in[16];
  const float* proj     = (const float*)d_in[17];
  float* outp = (float*)d_out;

  char* w = (char*)d_ws;
  size_t off = 0;
  auto alloc = [&](size_t bytes) {
    void* p = w + off;
    off = (off + bytes + 255) & ~(size_t)255;
    return p;
  };
  float* hf    = (float*)alloc((size_t)BN * CC * 4);
  u16* h0b     = (u16*)alloc((size_t)BN * CC * 2);
  u16* hb      = (u16*)alloc((size_t)BN * CC * 2);
  u16* xrb     = (u16*)alloc((size_t)BN * 64 * 2);
  float* kd    = (float*)alloc((size_t)BN * MM * 4);
  u16* ntb     = (u16*)alloc((size_t)BN * 128 * 2);
  u16* gb      = (u16*)kd;  // alias: kd+ntb (67MB) dead before gate phase (needs 50MB)
  u16* v1pb    = (u16*)alloc((size_t)BN * MM * 2);
  u16* v2pb    = (u16*)alloc((size_t)BN * MM * 2);
  float* v2x   = (float*)alloc((size_t)BB * MM * CC * 4);
  u16* v2xbT   = (u16*)alloc((size_t)BB * CC * MM * 2);
  float* v2s   = (float*)alloc((size_t)BB * MM * 4);
  u32* kmaxI   = (u32*)alloc(256);
  u16* Winb    = (u16*)alloc(64 * 64 * 2);
  u16* Wcatb   = (u16*)alloc(256 * 128 * 2);
  float* bcat  = (float*)alloc(256 * 4);
  u16* WiB     = (u16*)alloc((size_t)3 * CC * CC * 2);
  u16* WoB     = (u16*)alloc((size_t)3 * CC * CC * 2);
  u16* WregB   = (u16*)alloc(16 * 384 * 2);
  (void)ws_size; (void)n_in; (void)in_sizes; (void)out_size;

  // weight prep
  k_conv<<<108, 256, 0, stream>>>(Wi, WiB, 3 * CC * CC);
  k_conv<<<108, 256, 0, stream>>>(Wo, WoB, 3 * CC * CC);
  k_prep_win<<<16, 256, 0, stream>>>(W_in, Winb);
  k_prep_wcat<<<128, 256, 0, stream>>>(W1, W2, proj, b1, b2, Wcatb, bcat);
  k_prep_wreg<<<24, 256, 0, stream>>>(W_reg, WregB);
  k_init<<<1, 1024, 0, stream>>>(v2s, kmaxI);

  // embeddings + feature maps
  k_gather<<<BN / 256, 256, 0, stream>>>(x, node_emb, time_emb, xrb, ntb, hf, h0b, hb);
  k_inemb<<<1024, 256, 0, stream>>>(xrb, Winb, b_in, hf, h0b, hb);
  k_nvdd<<<2048, 128, 0, stream>>>(ntb, Wcatb, bcat, v1pb, kd, kmaxI);
  k_v2p<<<dim3(32, 16), 256, 0, stream>>>(kd, kmaxI, v2pb, v2s);

  // 3 linear-attention layers
  for (int l = 0; l < 3; ++l) {
    k_zero<<<192, 256, 0, stream>>>(v2x, BB * MM * CC);
    k_gate<<<1024, 256, 0, stream>>>(hb, WiB + (size_t)l * CC * CC,
                                     WoB + (size_t)l * CC * CC,
                                     bi + l * CC, bo + l * CC, gb);
    k_v2x<<<dim3(64, 16), 256, 0, stream>>>(v2pb, gb, v2x);
    k_trans<<<192, 256, 0, stream>>>(v2x, v2xbT);
    k_attnout<<<2048, 128, 0, stream>>>(v1pb, v2xbT, v2s, ln_g + l * CC,
                                        ln_b + l * CC, hf, hb);
  }

  // regression head
  k_head<<<1024, 256, 0, stream>>>(h0b, hb, WregB, b_reg, outp);
}
